// InContextAssoMemBlock_14534169329971
// MI455X (gfx1250) — matrix operand order repaired and checked by reading, not compiled, not run on hardware
//
#include <hip/hip_runtime.h>
#include <stdint.h>

// Problem constants (reference: B=2, T=2048, C=768, nh=12)
#define B_DIM 2
#define T_DIM 2048
#define C_DIM 768
#define NH    12
#define HS    64          // head size
#define KT    32          // key tile (== WMMA K for f16)
#define QT    64          // queries per block (4 waves x 16)

typedef _Float16 half_t;
typedef __attribute__((ext_vector_type(16))) _Float16 v16h;
typedef __attribute__((ext_vector_type(8)))  _Float16 v8h;
typedef __attribute__((ext_vector_type(8)))  float    v8f;
typedef __attribute__((ext_vector_type(4)))  unsigned int v4u;
typedef __attribute__((ext_vector_type(8)))  int          v8i;
typedef __attribute__((ext_vector_type(4)))  int          v4i;

// TDM builtin: only the ROCm 7.2 / clang-22 5-arg form has doc-verified
// semantics; on other toolchains use raw-ISA async global->LDS loads instead.
#if defined(__has_builtin)
#  if __has_builtin(__builtin_amdgcn_tensor_load_to_lds) && (__clang_major__ < 23)
#    define TDM_OK 1
#  endif
#endif
#ifndef TDM_OK
#  define TDM_OK 0
#endif

// ---------------------------------------------------------------------------
// WMMA fragment helpers (CDNA5 V_WMMA_F32_16X16X32_F16, wave32)
//
// A (16x32, M=lane%16):  elem e   -> K = 16*half + e   (16 contiguous)
// B (32x16, N=lane%16):  elem e   -> K = 16*half + e   (16 contiguous)
// C/D (16x16):           elem j   -> M = j + 8*half, N = lane%16
// ---------------------------------------------------------------------------
__device__ __forceinline__ v16h make_a_frag(const half_t* row_ptr, int hf) {
  v8h lo = *(const v8h*)(row_ptr + 16 * hf);
  v8h hi = *(const v8h*)(row_ptr + 16 * hf + 8);
  v16h a;
#pragma unroll
  for (int i = 0; i < 8; ++i) { a[i] = lo[i]; a[i + 8] = hi[i]; }
  return a;
}

__device__ __forceinline__ v16h make_b_frag(const half_t* p) {
  v8h lo = *(const v8h*)p;
  v8h hi = *(const v8h*)(p + 8);
  v16h b;
#pragma unroll
  for (int i = 0; i < 8; ++i) { b[i] = lo[i]; b[i + 8] = hi[i]; }
  return b;
}

__device__ __forceinline__ v8f wmma32(v16h a, v16h b, v8f c) {
  return __builtin_amdgcn_wmma_f32_16x16x32_f16(false, a, false, b, (short)0, c,
                                                false, false);
}

#if TDM_OK
// ---------------------------------------------------------------------------
// TDM descriptor: load a [KT=32 rows][64 f16] tile (contiguous rows) from
// global into LDS, padding +32B after every 128B row -> LDS pitch 160B.
// "Tensor" == the tile itself (dims 2..4 = 1) so OOB logic never triggers.
// ---------------------------------------------------------------------------
__device__ __forceinline__ void tdm_load_k_tile(unsigned lds_addr,
                                                const half_t* gptr) {
  unsigned long long ga = (unsigned long long)(uintptr_t)gptr;
  v4u g0;
  g0[0] = 1u;                                        // count=1, user descriptor
  g0[1] = lds_addr;                                  // LDS byte address
  g0[2] = (unsigned)ga;                              // global_addr[31:0]
  g0[3] = (unsigned)((ga >> 32) & 0x01FFFFFFu)       // global_addr[56:32]
          | (2u << 30);                              // type = 2 ("image")
  v8i g1;
  g1[0] = (int)((1u << 16)      // data_size = 1 -> 2 bytes
              | (1u << 20)      // pad_enable
              | (4u << 22)      // pad_interval: 32 DWORDs (one 128B row)
              | (7u << 25));    // pad_amount:   8 DWORDs (32B)
  g1[1] = (int)(64u << 16);     // tensor_dim0[15:0] = 64
  g1[2] = (int)(32u << 16);     // tensor_dim1 = 32
  g1[3] = (int)(64u << 16);     // tile_dim0 = 64
  g1[4] = 32;                   // tile_dim1 = 32, tile_dim2 = 0 (unused)
  g1[5] = 64;                   // tensor_dim0_stride = 64 elements
  g1[6] = 0;
  g1[7] = 0;
  v4i g2;
  g2[0] = 1;                    // tensor_dim2 = 1
  g2[1] = 1;                    // tensor_dim3 = 1
  g2[2] = 0;                    // tensor_dim2_stride
  g2[3] = 0;                    // tile_dim3 = 0 (unused)
  v4i g3;
  g3[0] = 0;                    // tensor_dim3_stride
  g3[1] = (int)(1u << 16);      // tensor_dim4 = 1
  g3[2] = 0;
  g3[3] = 0;                    // tile_dim4 = 0 (unused)
  __builtin_amdgcn_tensor_load_to_lds(g0, g1, g2, g3, 0);
}
#endif

// ---------------------------------------------------------------------------
// 0) f32 -> f16 conversion
// ---------------------------------------------------------------------------
__global__ void f32_to_f16_kernel(const float* __restrict__ in,
                                  half_t* __restrict__ out, int n) {
  int i = blockIdx.x * blockDim.x + threadIdx.x;
  if (i < n) out[i] = (half_t)in[i];
}

// ---------------------------------------------------------------------------
// 1) GEMM: C[m,n] = sum_k A[m,k] * W[n,k]    (A: MxK f16, W: NxK f16, C f32)
//    block = 4 waves; each wave -> 16x64 tile; block -> 64x64 tile
// ---------------------------------------------------------------------------
__global__ __launch_bounds__(128) void gemm_nt_kernel(
    const half_t* __restrict__ A, const half_t* __restrict__ W,
    float* __restrict__ Cout, int M, int N, int K) {
  const int w    = threadIdx.x >> 5;
  const int lane = threadIdx.x & 31;
  const int l16  = lane & 15;
  const int hf   = lane >> 4;

  const int mbase = blockIdx.y * 64 + w * 16;
  const int nbase = blockIdx.x * 64;

  const half_t* Arow = A + (size_t)(mbase + l16) * K;
  const half_t* W0 = W + (size_t)(nbase + 0  + l16) * K + 16 * hf;
  const half_t* W1 = W + (size_t)(nbase + 16 + l16) * K + 16 * hf;
  const half_t* W2 = W + (size_t)(nbase + 32 + l16) * K + 16 * hf;
  const half_t* W3 = W + (size_t)(nbase + 48 + l16) * K + 16 * hf;

  v8f acc0 = {}, acc1 = {}, acc2 = {}, acc3 = {};
  for (int kb = 0; kb < K; kb += 32) {
    v16h a  = make_a_frag(Arow + kb, hf);
    v16h b0 = make_b_frag(W0 + kb);
    v16h b1 = make_b_frag(W1 + kb);
    v16h b2 = make_b_frag(W2 + kb);
    v16h b3 = make_b_frag(W3 + kb);
    acc0 = wmma32(a, b0, acc0);
    acc1 = wmma32(a, b1, acc1);
    acc2 = wmma32(a, b2, acc2);
    acc3 = wmma32(a, b3, acc3);
  }
#pragma unroll
  for (int j = 0; j < 8; ++j) {
    size_t row = (size_t)(mbase + j + 8 * hf) * N + nbase + l16;
    Cout[row + 0]  = acc0[j];
    Cout[row + 16] = acc1[j];
    Cout[row + 32] = acc2[j];
    Cout[row + 48] = acc3[j];
  }
}

// ---------------------------------------------------------------------------
// 2) Leaky average (serial EMA over T), in place: u -> k0
// ---------------------------------------------------------------------------
__global__ void leaky_avg_kernel(float* __restrict__ u,
                                 const float* __restrict__ la_coef) {
  int idx = blockIdx.x * blockDim.x + threadIdx.x;  // [B*C]
  if (idx >= B_DIM * C_DIM) return;
  int b = idx / C_DIM, c = idx % C_DIM;
  int h = c / HS;
  float cc = la_coef[h];
  float omc = 1.0f - cc;
  float* p = u + (size_t)b * T_DIM * C_DIM + c;
  float y = 0.0f;
  for (int t = 0; t < T_DIM; ++t) {
    y = cc * y + omc * p[(size_t)t * C_DIM];
    p[(size_t)t * C_DIM] = y;
  }
}

// ---------------------------------------------------------------------------
// 3) normalize k0 rows, scale by exp(min(kernel_beta*10,5)), write f16
// ---------------------------------------------------------------------------
__global__ void normalize_k_kernel(const float* __restrict__ k0,
                                   const float* __restrict__ kernel_beta,
                                   half_t* __restrict__ kh) {
  int wave = (blockIdx.x * blockDim.x + threadIdx.x) >> 5;
  int lane = threadIdx.x & 31;
  int total = B_DIM * NH * T_DIM;
  if (wave >= total) return;
  int t = wave % T_DIM;
  int h = (wave / T_DIM) % NH;
  int b = wave / (T_DIM * NH);

  const float* src = k0 + ((size_t)b * T_DIM + t) * C_DIM + h * HS;
  float x0 = src[lane], x1 = src[lane + 32];
  float ss = x0 * x0 + x1 * x1;
#pragma unroll
  for (int m = 16; m >= 1; m >>= 1) ss += __shfl_xor(ss, m, 32);
  float beta = __expf(fminf(kernel_beta[h] * 10.0f, 5.0f));
  float inv = beta / (sqrtf(ss) + 1e-6f);
  half_t* dst = kh + (((size_t)b * NH + h) * T_DIM + t) * HS;
  dst[lane]      = (half_t)(x0 * inv);
  dst[lane + 32] = (half_t)(x1 * inv);
}

// ---------------------------------------------------------------------------
// 4) v = normalize(shift(xl)*(1-vc) + xl*vc) * exp(min(value_beta*10,5))
// ---------------------------------------------------------------------------
__global__ void make_v_kernel(const float* __restrict__ xl,
                              const float* __restrict__ v_coef,
                              const float* __restrict__ value_beta,
                              half_t* __restrict__ vh) {
  int wave = (blockIdx.x * blockDim.x + threadIdx.x) >> 5;
  int lane = threadIdx.x & 31;
  int total = B_DIM * NH * T_DIM;
  if (wave >= total) return;
  int t = wave % T_DIM;
  int h = (wave / T_DIM) % NH;
  int b = wave / (T_DIM * NH);

  const float* cur = xl + ((size_t)b * T_DIM + t) * C_DIM + h * HS;
  float c0 = cur[lane], c1 = cur[lane + 32];
  float s0 = 0.0f, s1 = 0.0f;
  if (t + 1 < T_DIM) {
    const float* nxt = cur + C_DIM;
    s0 = nxt[lane]; s1 = nxt[lane + 32];
  }
  float vc = v_coef[h];
  float f0 = s0 * (1.0f - vc) + c0 * vc;
  float f1 = s1 * (1.0f - vc) + c1 * vc;
  float ss = f0 * f0 + f1 * f1;
#pragma unroll
  for (int m = 16; m >= 1; m >>= 1) ss += __shfl_xor(ss, m, 32);
  float vb = __expf(fminf(value_beta[h] * 10.0f, 5.0f));
  float inv = vb / (sqrtf(ss) + 1e-6f);
  half_t* dst = vh + (((size_t)b * NH + h) * T_DIM + t) * HS;
  dst[lane]      = (half_t)(f0 * inv);
  dst[lane + 32] = (half_t)(f1 * inv);
}

// ---------------------------------------------------------------------------
// 5) zero rows t==0 of attn output (y[:,0,:] = 0 in the reference)
// ---------------------------------------------------------------------------
__global__ void zero_t0_kernel(half_t* __restrict__ attnh) {
  int i = blockIdx.x * blockDim.x + threadIdx.x;
  if (i >= B_DIM * C_DIM) return;
  attnh[(size_t)(i / C_DIM) * T_DIM * C_DIM + (i % C_DIM)] = (half_t)0.0f;
}

// ---------------------------------------------------------------------------
// 6) Flash-attention: q = k[1:], causal (t_k < t_q), softmax, o = P @ v
//    grid = (ceil((T-1)/QT), B*nh), block = 128 (4 waves, 16 queries each)
//    K tile staged by async DMA to LDS (TDM on clang-22, raw-ISA
//    global_load_async_to_lds_b128 otherwise); V tile staged transposed.
// ---------------------------------------------------------------------------
__global__ __launch_bounds__(128) void attention_kernel(
    const half_t* __restrict__ kh, const half_t* __restrict__ vh,
    half_t* __restrict__ attnh) {
  const int bh   = blockIdx.y;
  const int b    = bh / NH;
  const int h    = bh % NH;
  const int w    = threadIdx.x >> 5;
  const int lane = threadIdx.x & 31;
  const int l16  = lane & 15;
  const int hf   = lane >> 4;

  const int Tm1  = T_DIM - 1;
  const int qb0  = 1 + QT * blockIdx.x;                       // first query time
  const int qmax = (qb0 + QT - 1 < Tm1) ? (qb0 + QT - 1) : Tm1;
  const int ntiles = (qmax + KT - 1) / KT;                    // keys 0..qmax-1

  const half_t* Kp = kh + (size_t)bh * T_DIM * HS;
  const half_t* Vp = vh + (size_t)bh * T_DIM * HS;

  __shared__ __align__(32) half_t ldsK[KT][80];      // [key][dim], 160B pitch
  __shared__ __align__(32) half_t ldsVt[HS][48];     // [dim][key], padded
  __shared__ __align__(32) half_t ldsP[4][16][40];   // per-wave P transpose tile

  const unsigned ldsK_off = (unsigned)(uintptr_t)&ldsK[0][0];  // LDS byte addr

  // Q fragments (Q rows are k rows at time t_q); clamp OOB rows (masked later)
  int tq_row = qb0 + w * 16 + l16;
  if (tq_row > Tm1) tq_row = Tm1;
  const half_t* Qrow = Kp + (size_t)tq_row * HS;
  v16h qa0 = make_a_frag(Qrow, hf);        // dims 0..31
  v16h qa1 = make_a_frag(Qrow + 32, hf);   // dims 32..63

  v8f acc0 = {}, acc1 = {}, acc2 = {}, acc3 = {};
  float mrow[8], lrow[8];
#pragma unroll
  for (int j = 0; j < 8; ++j) { mrow[j] = -1e30f; lrow[j] = 0.0f; }

  for (int kt = 0; kt < ntiles; ++kt) {
    const int kbase = kt * KT;
    __syncthreads();

#if TDM_OK
    // K tile via Tensor Data Mover (wave 0 only; TDM ignores EXEC so the
    // issue must be wave-granular). Overlaps with V staging below.
    if (threadIdx.x < 32) {
      tdm_load_k_tile(ldsK_off, Kp + (size_t)kbase * HS);
    }
#else
    // K tile via per-lane async global->LDS DMA (ASYNCcnt-tracked).
    // 32 rows x 128B = 256 x 16B chunks; LDS row pitch = 160B (padded).
    for (int idx = threadIdx.x; idx < KT * 8; idx += blockDim.x) {
      int r = idx >> 3, c = idx & 7;
      unsigned lds_addr = ldsK_off + (unsigned)(r * 160 + c * 16);
      unsigned long long ga =
          (unsigned long long)(uintptr_t)(Kp + (size_t)(kbase + r) * HS + c * 8);
      asm volatile("global_load_async_to_lds_b128 %0, %1, off"
                   :: "v"(lds_addr), "v"(ga)
                   : "memory");
    }
#endif
    // prefetch next tiles into cache while this one is staged
    if (kt + 1 < ntiles) {
      __builtin_prefetch(Kp + (size_t)(kbase + KT) * HS, 0, 0);
      __builtin_prefetch(Vp + (size_t)(kbase + KT) * HS, 0, 0);
    }
    // stage V tile transposed [HS][KT]
    for (int idx = threadIdx.x; idx < KT * HS / 2; idx += blockDim.x) {
      int r = idx / (HS / 2), c2 = idx % (HS / 2);
      uint32_t uv = *(const uint32_t*)&Vp[(size_t)(kbase + r) * HS + c2 * 2];
      half_t h0 = ((const half_t*)&uv)[0];
      half_t h1 = ((const half_t*)&uv)[1];
      ldsVt[c2 * 2][r]     = h0;
      ldsVt[c2 * 2 + 1][r] = h1;
    }
#if TDM_OK
    if (threadIdx.x < 32) {
      __builtin_amdgcn_s_wait_tensorcnt((short)0);   // DMA done writing LDS
    }
#else
    asm volatile("s_wait_asynccnt 0x0" ::: "memory"); // per-wave async done
#endif
    __syncthreads();

    // S = Q K^T  (16 queries x 32 keys, two 16-key groups)
    v16h b00 = make_b_frag(&ldsK[l16][0  + 16 * hf]);
    v16h b01 = make_b_frag(&ldsK[l16][32 + 16 * hf]);
    v16h b10 = make_b_frag(&ldsK[16 + l16][0  + 16 * hf]);
    v16h b11 = make_b_frag(&ldsK[16 + l16][32 + 16 * hf]);
    v8f s0 = {}, s1 = {};
    s0 = wmma32(qa0, b00, s0);
    s0 = wmma32(qa1, b01, s0);
    s1 = wmma32(qa0, b10, s1);
    s1 = wmma32(qa1, b11, s1);

    // online softmax + write P (f16) to per-wave LDS transpose tile
#pragma unroll
    for (int j = 0; j < 8; ++j) {
      int tq = qb0 + w * 16 + j + 8 * hf;
      float a = s0[j], c = s1[j];
      if (kbase + l16 >= tq      || tq > Tm1) a = -1e30f;
      if (kbase + 16 + l16 >= tq || tq > Tm1) c = -1e30f;
      float r = fmaxf(a, c);
      r = fmaxf(r, __shfl_xor(r, 1, 32));
      r = fmaxf(r, __shfl_xor(r, 2, 32));
      r = fmaxf(r, __shfl_xor(r, 4, 32));
      r = fmaxf(r, __shfl_xor(r, 8, 32));
      float mn = fmaxf(mrow[j], r);
      float sc = __expf(mrow[j] - mn);
      float p0 = __expf(a - mn);
      float p1 = __expf(c - mn);
      float rs = p0 + p1;
      rs += __shfl_xor(rs, 1, 32);
      rs += __shfl_xor(rs, 2, 32);
      rs += __shfl_xor(rs, 4, 32);
      rs += __shfl_xor(rs, 8, 32);
      lrow[j] = lrow[j] * sc + rs;
      mrow[j] = mn;
      acc0[j] *= sc; acc1[j] *= sc; acc2[j] *= sc; acc3[j] *= sc;
      ldsP[w][j + 8 * hf][l16]      = (half_t)p0;
      ldsP[w][j + 8 * hf][16 + l16] = (half_t)p1;
    }
    // intra-wave LDS RAW fence (stores above feed the A-frag loads below)
    asm volatile("s_wait_dscnt 0x0" ::: "memory");

    // O += P @ V   (P: 16x32 f16 A-frag, V: 32x64 via 4 B-frags)
    v16h pa  = make_a_frag(&ldsP[w][l16][0], hf);
    v16h vb0 = make_b_frag(&ldsVt[0  + l16][16 * hf]);
    v16h vb1 = make_b_frag(&ldsVt[16 + l16][16 * hf]);
    v16h vb2 = make_b_frag(&ldsVt[32 + l16][16 * hf]);
    v16h vb3 = make_b_frag(&ldsVt[48 + l16][16 * hf]);
    acc0 = wmma32(pa, vb0, acc0);
    acc1 = wmma32(pa, vb1, acc1);
    acc2 = wmma32(pa, vb2, acc2);
    acc3 = wmma32(pa, vb3, acc3);
  }

  // epilogue: divide by softmax denom, store f16 into [B,T,C] at row t_q
#pragma unroll
  for (int j = 0; j < 8; ++j) {
    int tq = qb0 + w * 16 + j + 8 * hf;
    if (tq > Tm1) continue;
    float inv = 1.0f / lrow[j];
    size_t base = ((size_t)b * T_DIM + tq) * C_DIM + h * HS + l16;
    attnh[base + 0]  = (half_t)(acc0[j] * inv);
    attnh[base + 16] = (half_t)(acc1[j] * inv);
    attnh[base + 32] = (half_t)(acc2[j] * inv);
    attnh[base + 48] = (half_t)(acc3[j] * inv);
  }
}

// ---------------------------------------------------------------------------
// launcher
// ---------------------------------------------------------------------------
extern "C" void kernel_launch(void* const* d_in, const int* in_sizes, int n_in,
                              void* d_out, int out_size, void* d_ws,
                              size_t ws_size, hipStream_t stream) {
  const float* x           = (const float*)d_in[0];
  const float* W_la        = (const float*)d_in[1];
  const float* la_coef     = (const float*)d_in[2];
  const float* W_v         = (const float*)d_in[3];
  const float* v_coef      = (const float*)d_in[4];
  const float* kernel_beta = (const float*)d_in[5];
  const float* value_beta  = (const float*)d_in[6];
  const float* W_proj      = (const float*)d_in[7];
  float* out = (float*)d_out;

  const size_t MT = (size_t)B_DIM * T_DIM;  // 4096 rows
  const size_t NX = MT * C_DIM;             // elements in x-like tensors
  const size_t NW = (size_t)C_DIM * C_DIM;

  char* ws = (char*)d_ws;
  half_t* xh     = (half_t*)ws; ws += NX * sizeof(half_t);
  half_t* attnh  = xh;  // reuse: xh is dead before attention writes begin
  half_t* wlah   = (half_t*)ws; ws += NW * sizeof(half_t);
  half_t* wvh    = (half_t*)ws; ws += NW * sizeof(half_t);
  half_t* wprojh = (half_t*)ws; ws += NW * sizeof(half_t);
  float*  u      = (float*)ws;  ws += NX * sizeof(float);   // becomes k0 in place
  float*  xl     = (float*)ws;  ws += NX * sizeof(float);
  half_t* khbuf  = (half_t*)ws; ws += NX * sizeof(half_t);  // [B,nh,T,HS]
  half_t* vhbuf  = (half_t*)ws; ws += NX * sizeof(half_t);  // [B,nh,T,HS]

  // 0) f32 -> f16 conversions
  f32_to_f16_kernel<<<(int)((NX + 255) / 256), 256, 0, stream>>>(x, xh, (int)NX);
  f32_to_f16_kernel<<<(int)((NW + 255) / 256), 256, 0, stream>>>(W_la, wlah, (int)NW);
  f32_to_f16_kernel<<<(int)((NW + 255) / 256), 256, 0, stream>>>(W_v, wvh, (int)NW);
  f32_to_f16_kernel<<<(int)((NW + 255) / 256), 256, 0, stream>>>(W_proj, wprojh, (int)NW);

  // 1) u = x @ W_la^T ; xl = x @ W_v^T
  dim3 ggrid(C_DIM / 64, (unsigned)(MT / 64));
  gemm_nt_kernel<<<ggrid, 128, 0, stream>>>(xh, wlah, u, (int)MT, C_DIM, C_DIM);
  gemm_nt_kernel<<<ggrid, 128, 0, stream>>>(xh, wvh, xl, (int)MT, C_DIM, C_DIM);

  // 2) serial EMA over time (in place)
  leaky_avg_kernel<<<(B_DIM * C_DIM + 255) / 256, 256, 0, stream>>>(u, la_coef);

  // 3/4) normalize + scale -> f16 k and v
  int rows = B_DIM * NH * T_DIM;  // one wave per row, 8 waves per block
  normalize_k_kernel<<<rows / 8, 256, 0, stream>>>(u, kernel_beta, khbuf);
  make_v_kernel<<<rows / 8, 256, 0, stream>>>(xl, v_coef, value_beta, vhbuf);

  // 5) zero t==0 output rows, then attention
  zero_t0_kernel<<<(B_DIM * C_DIM + 255) / 256, 256, 0, stream>>>(attnh);
  dim3 agrid((T_DIM - 1 + QT - 1) / QT, B_DIM * NH);
  attention_kernel<<<agrid, 128, 0, stream>>>(khbuf, vhbuf, attnh);

  // 6) out = attn @ W_proj^T
  gemm_nt_kernel<<<ggrid, 128, 0, stream>>>(attnh, wprojh, out, (int)MT, C_DIM, C_DIM);
}
